// WeightSharedBlock_60627758350901
// MI455X (gfx1250) — matrix operand order repaired and checked by reading, not compiled, not run on hardware
//
#include <hip/hip_runtime.h>
#include <hip/hip_bf16.h>

// ---------------------------------------------------------------------------
// Types for CDNA5 WMMA (wave32): bf16 16x16x32, f32 accumulate.
// ---------------------------------------------------------------------------
typedef __bf16          v16bf __attribute__((ext_vector_type(16)));
typedef unsigned short  v16u  __attribute__((ext_vector_type(16)));
typedef unsigned short  v8u   __attribute__((ext_vector_type(8)));
typedef float           v8f   __attribute__((ext_vector_type(8)));

static constexpr int Bn   = 4;
static constexpr int Nn   = 1024;
static constexpr int Cc   = 768;
static constexpr int Hh   = 12;
static constexpr int HD   = 64;
static constexpr int DFF  = 3072;
static constexpr int TOK  = Bn * Nn;      // 4096 tokens

// Hardware bf16 convert (gfx1250 native cvt, RNE).
__device__ __forceinline__ unsigned short f2bf(float f) {
  __bf16 h = (__bf16)f;
  return __builtin_bit_cast(unsigned short, h);
}
__device__ __forceinline__ float bf2f(unsigned short h) {
  unsigned u = ((unsigned)h) << 16;
  return __builtin_bit_cast(float, u);
}

// Async global -> LDS copy of 16 bytes per lane (CDNA5, tracked by ASYNCcnt).
__device__ __forceinline__ void async_b128(const unsigned short* g,
                                           const unsigned short* l) {
  unsigned loff = (unsigned)(size_t)l;   // low 32 bits = LDS byte offset
  asm volatile("global_load_async_to_lds_b128 %0, %1, off"
               :: "v"(loff), "v"(g) : "memory");
}
__device__ __forceinline__ void wait_async0() {
  asm volatile("s_wait_asynccnt 0" ::: "memory");
}

// ---------------------------------------------------------------------------
// WMMA GEMM body:  C[M,N] = A[M,K](bf16) @ W[N,K](bf16)^T  (+epilogue)
// Block = 256 threads = 8 waves (WR x WC); each wave computes a
// (TM*16)x(TN*16) tile of TMxTN WMMA 16x16x32 tiles.  K-slabs of 32 are
// double-buffered in LDS via global_load_async_to_lds_b128, overlapping the
// DMA of slab i+1 with WMMAs on slab i.  LDS rows padded to 40 shorts (80B).
//
// Fragment layouts per CDNA5 ISA (wave32):
//  A 16x32 bf16 : lane l (l<16) row=l holds K {0..7,16..23}; lane l+16 holds
//                 K {8..15,24..31}  -> two 16B LDS reads per lane.
//  B 32x16 bf16 : lane l (l<16) col=l holds K {0..7,16..23}; lane l+16 holds
//                 K {8..15,24..31} (same as A) -> two 16B reads.
//  C/D 16x16 f32: vgpr v, lane l -> (m = v + (l>=16)*8, n = l%16).
// ---------------------------------------------------------------------------
template <int WR, int WC, int TM, int TN, int EPI>
__device__ __forceinline__ void gemm_body(
    const unsigned short* __restrict__ A,  int lda,
    const unsigned short* __restrict__ Wm, int ldb,
    void* __restrict__ Cv, int ldc, int K,
    const float* __restrict__ bias,
    const float* __restrict__ res, int ldres,
    float scale)
{
  constexpr int BM = WR * TM * 16;
  constexpr int BN = WC * TN * 16;
  constexpr int LR = 40;                       // padded LDS row, in shorts
  __shared__ unsigned short sA[2][BM * LR];
  __shared__ unsigned short sB[2][BN * LR];

  const int tid  = threadIdx.x;
  const int lane = tid & 31;
  const int wave = tid >> 5;
  const int wr   = wave / WC;
  const int wc   = wave % WC;
  const int mbB  = blockIdx.y * BM;
  const int nbB  = blockIdx.x * BN;
  const int lm   = lane & 15;
  const int lh   = lane >> 4;

  v8f acc[TM][TN];
#pragma unroll
  for (int i = 0; i < TM; ++i)
#pragma unroll
    for (int j = 0; j < TN; ++j) acc[i][j] = {};

  auto issue = [&](int buf, int k0) {
#pragma unroll
    for (int idx = tid; idx < BM * 4; idx += 256) {       // A: BM rows x 64B
      const int row = idx >> 2, ch = idx & 3;
      async_b128(A + (size_t)(mbB + row) * lda + k0 + ch * 8,
                 &sA[buf][row * LR + ch * 8]);
    }
#pragma unroll
    for (int idx = tid; idx < BN * 4; idx += 256) {       // B: BN rows x 64B
      const int row = idx >> 2, ch = idx & 3;
      async_b128(Wm + (size_t)(nbB + row) * ldb + k0 + ch * 8,
                 &sB[buf][row * LR + ch * 8]);
    }
  };

  issue(0, 0);
  const int nk = K / 32;
  for (int i = 0; i < nk; ++i) {
    const int buf = i & 1;
    wait_async0();          // our slab-i async ops done
    __syncthreads();        // -> everyone's are done; buf^1 free to overwrite
    if (i + 1 < nk) issue(buf ^ 1, (i + 1) * 32);

    v16u af[TM], bfr[TN];
#pragma unroll
    for (int tm = 0; tm < TM; ++tm) {
      const unsigned short* ap =
          &sA[buf][(wr * TM * 16 + tm * 16 + lm) * LR + lh * 8];
      v8u lo = *(const v8u*)ap;          // K = lh*8 .. +7
      v8u hi = *(const v8u*)(ap + 16);   // K = 16 + lh*8 .. +7
      v16u t;
#pragma unroll
      for (int e = 0; e < 8; ++e) { t[e] = lo[e]; t[e + 8] = hi[e]; }
      af[tm] = t;
    }
#pragma unroll
    for (int tn = 0; tn < TN; ++tn) {
      const unsigned short* bp =
          &sB[buf][(wc * TN * 16 + tn * 16 + lm) * LR + lh * 8];
      v8u lo = *(const v8u*)bp;
      v8u hi = *(const v8u*)(bp + 16);
      v16u t;
#pragma unroll
      for (int e = 0; e < 8; ++e) { t[e] = lo[e]; t[e + 8] = hi[e]; }
      bfr[tn] = t;
    }
#pragma unroll
    for (int tm = 0; tm < TM; ++tm)
#pragma unroll
      for (int tn = 0; tn < TN; ++tn)
        acc[tm][tn] = __builtin_amdgcn_wmma_f32_16x16x32_bf16(
            false, __builtin_bit_cast(v16bf, af[tm]),
            false, __builtin_bit_cast(v16bf, bfr[tn]),
            (short)0, acc[tm][tn], false, false);
  }

#pragma unroll
  for (int tm = 0; tm < TM; ++tm)
#pragma unroll
    for (int tn = 0; tn < TN; ++tn) {
      const int n = nbB + wc * TN * 16 + tn * 16 + lm;
#pragma unroll
      for (int v = 0; v < 8; ++v) {
        const int m = mbB + wr * TM * 16 + tm * 16 + lh * 8 + v;
        const float val = acc[tm][tn][v];
        if constexpr (EPI == 0) {          // bf16, scaled (attention scores)
          ((unsigned short*)Cv)[(size_t)m * ldc + n] = f2bf(val * scale);
        } else if constexpr (EPI == 1) {   // bf16, +bias (qkv)
          ((unsigned short*)Cv)[(size_t)m * ldc + n] = f2bf(val + bias[n]);
        } else if constexpr (EPI == 2) {   // bf16 plain (attn @ v)
          ((unsigned short*)Cv)[(size_t)m * ldc + n] = f2bf(val);
        } else if constexpr (EPI == 3) {   // f32, +bias +residual (proj, fc2)
          ((float*)Cv)[(size_t)m * ldc + n] =
              val + bias[n] + res[(size_t)m * ldres + n];
        } else if constexpr (EPI == 4) {   // bf16, GELU(bias+x) (fc1)
          const float x = val + bias[n];
          ((unsigned short*)Cv)[(size_t)m * ldc + n] =
              f2bf(0.5f * x * (1.0f + erff(x * 0.70710678118f)));
        }
      }
    }
}

template <int WR, int WC, int TM, int TN, int EPI>
__global__ __launch_bounds__(256) void k_gemm(
    const unsigned short* A, int lda, const unsigned short* Wm, int ldb,
    void* C, int ldc, int K, const float* bias, const float* res, int ldres,
    float scale)
{
  gemm_body<WR, WC, TM, TN, EPI>(A, lda, Wm, ldb, C, ldc, K, bias, res,
                                 ldres, scale);
}

// Batched scores: per z = b*12+h,  S = (q @ k^T) * 1/sqrt(HD), bf16 out.
__global__ __launch_bounds__(256) void k_gemm_scores(
    const unsigned short* qkv, unsigned short* Sc)
{
  const int z = blockIdx.z, b = z / Hh, h = z % Hh;
  const unsigned short* A  = qkv + (size_t)b * Nn * (3 * Cc) + h * HD;
  const unsigned short* Wm = qkv + (size_t)b * Nn * (3 * Cc) + Cc + h * HD;
  unsigned short* C = Sc + ((size_t)z << 20);
  gemm_body<2, 4, 2, 4, 0>(A, 3 * Cc, Wm, 3 * Cc, C, Nn, HD,
                           nullptr, nullptr, 0, 0.125f);
}

// Batched attn @ v: per z, A = probs [1024,1024], B = vT [64,1024]; output
// scattered into token-major [tok, g*64+d] layout.
__global__ __launch_bounds__(256) void k_gemm_av(
    const unsigned short* P, const unsigned short* vT, unsigned short* AO)
{
  const int z = blockIdx.z, b = z / Hh, g = z % Hh;
  const unsigned short* A  = P  + ((size_t)z << 20);
  const unsigned short* Wm = vT + (size_t)z * HD * Nn;
  unsigned short* C = AO + (size_t)b * Nn * Cc + g * HD;
  gemm_body<4, 2, 2, 2, 2>(A, Nn, Wm, Nn, C, Cc, Nn, nullptr, nullptr, 0,
                           1.0f);
}

// ---------------------------------------------------------------------------
// LayerNorm: one wave32 per token (C=768 -> 24 elems/lane).
// ---------------------------------------------------------------------------
template <typename OutT>
__global__ __launch_bounds__(256) void k_ln(
    const float* __restrict__ X, const float* __restrict__ G,
    const float* __restrict__ Be, OutT* __restrict__ Y)
{
  const int wv = threadIdx.x >> 5, lane = threadIdx.x & 31;
  const size_t t = (size_t)blockIdx.x * 8 + wv;
  const float* xr = X + t * Cc;
  float vals[24], sum = 0.f, sq = 0.f;
#pragma unroll
  for (int i = 0; i < 24; ++i) {
    float v = xr[lane + i * 32];
    vals[i] = v; sum += v; sq += v * v;
  }
#pragma unroll
  for (int o = 16; o > 0; o >>= 1) {
    sum += __shfl_xor(sum, o, 32);
    sq  += __shfl_xor(sq,  o, 32);
  }
  const float mu  = sum * (1.0f / Cc);
  const float inv = rsqrtf(sq * (1.0f / Cc) - mu * mu + 1e-5f);
#pragma unroll
  for (int i = 0; i < 24; ++i) {
    const int c = lane + i * 32;
    const float yv = (vals[i] - mu) * inv * G[c] + Be[c];
    if constexpr (sizeof(OutT) == 2) Y[t * Cc + c] = f2bf(yv);
    else                             Y[t * Cc + c] = yv;
  }
}

// ---------------------------------------------------------------------------
// Depthwise conv (K=7 over tokens, zero pad) + bias + LayerNorm(gm,bm) -> bf16
// One wave per token.
// ---------------------------------------------------------------------------
__global__ __launch_bounds__(256) void k_convln(
    const float* __restrict__ H2, const float* __restrict__ W7,
    const float* __restrict__ Wb, const float* __restrict__ Gm,
    const float* __restrict__ Bm, unsigned short* __restrict__ out)
{
  const int wv = threadIdx.x >> 5, lane = threadIdx.x & 31;
  const int t = blockIdx.x * 8 + wv;
  const int b = t >> 10, n = t & 1023;
  float vals[24], sum = 0.f, sq = 0.f;
#pragma unroll
  for (int i = 0; i < 24; ++i) {
    const int c = lane + i * 32;
    float acc = Wb[c];
#pragma unroll
    for (int j = 0; j < 7; ++j) {
      const int nn = n + j - 3;
      if (nn >= 0 && nn < Nn)
        acc += H2[((size_t)(b * Nn + nn)) * Cc + c] * W7[c * 7 + j];
    }
    vals[i] = acc; sum += acc; sq += acc * acc;
  }
#pragma unroll
  for (int o = 16; o > 0; o >>= 1) {
    sum += __shfl_xor(sum, o, 32);
    sq  += __shfl_xor(sq,  o, 32);
  }
  const float mu  = sum * (1.0f / Cc);
  const float inv = rsqrtf(sq * (1.0f / Cc) - mu * mu + 1e-5f);
#pragma unroll
  for (int i = 0; i < 24; ++i) {
    const int c = lane + i * 32;
    out[(size_t)t * Cc + c] = f2bf((vals[i] - mu) * inv * Gm[c] + Bm[c]);
  }
}

// ---------------------------------------------------------------------------
// Head-mix(T_before) -> softmax over m -> head-mix(T_after). One block per
// (b,n) row; all 12 heads' score rows staged in LDS (~50 KB of 320 KB WGP LDS)
// ---------------------------------------------------------------------------
__global__ __launch_bounds__(256) void k_mixsoftmax(
    const unsigned short* __restrict__ S, const float* __restrict__ Tb,
    const float* __restrict__ Ta, unsigned short* __restrict__ P)
{
  __shared__ unsigned short sS[Hh * Nn];
  __shared__ unsigned short sP[Hh * Nn];
  __shared__ float tb[Hh * Hh], ta[Hh * Hh];
  __shared__ float red[8];

  const int tid = threadIdx.x;
  const int bi = blockIdx.x, b = bi >> 10, n = bi & 1023;

  for (int h = 0; h < Hh; ++h) {
    const size_t gb = (((size_t)(b * Hh + h)) << 20) + ((size_t)n << 10);
    for (int m = tid; m < Nn; m += 256) sS[h * Nn + m] = S[gb + m];
  }
  if (tid < Hh * Hh) { tb[tid] = Tb[tid]; ta[tid] = Ta[tid]; }
  __syncthreads();

  // mix before softmax: sP[g][m] = sum_h Tb[g][h] * sS[h][m]
  for (int idx = tid; idx < Hh * Nn; idx += 256) {
    const int g = idx >> 10, m = idx & 1023;
    float a = 0.f;
#pragma unroll
    for (int h = 0; h < Hh; ++h) a += tb[g * Hh + h] * bf2f(sS[h * Nn + m]);
    sP[idx] = f2bf(a);
  }
  __syncthreads();

  // softmax each row g over m
  const int lane = tid & 31, wv = tid >> 5;
  for (int g = 0; g < Hh; ++g) {
    float mx = -3.4e38f;
    for (int m = tid; m < Nn; m += 256) mx = fmaxf(mx, bf2f(sP[g * Nn + m]));
#pragma unroll
    for (int o = 16; o > 0; o >>= 1) mx = fmaxf(mx, __shfl_xor(mx, o, 32));
    if (lane == 0) red[wv] = mx;
    __syncthreads();
    if (tid == 0) {
      float v = red[0];
      for (int i = 1; i < 8; ++i) v = fmaxf(v, red[i]);
      red[0] = v;
    }
    __syncthreads();
    mx = red[0];
    float sum = 0.f;
    for (int m = tid; m < Nn; m += 256) {
      const float e = __expf(bf2f(sP[g * Nn + m]) - mx);
      sP[g * Nn + m] = f2bf(e);
      sum += e;
    }
#pragma unroll
    for (int o = 16; o > 0; o >>= 1) sum += __shfl_xor(sum, o, 32);
    if (lane == 0) red[wv] = sum;
    __syncthreads();
    if (tid == 0) {
      float v = 0.f;
      for (int i = 0; i < 8; ++i) v += red[i];
      red[0] = v;
    }
    __syncthreads();
    const float inv = 1.0f / red[0];
    for (int m = tid; m < Nn; m += 256)
      sP[g * Nn + m] = f2bf(bf2f(sP[g * Nn + m]) * inv);
    __syncthreads();
  }

  // mix after softmax + store bf16
  for (int idx = tid; idx < Hh * Nn; idx += 256) {
    const int g = idx >> 10, m = idx & 1023;
    float a = 0.f;
#pragma unroll
    for (int h = 0; h < Hh; ++h) a += ta[g * Hh + h] * bf2f(sP[h * Nn + m]);
    const size_t gb = (((size_t)(b * Hh + g)) << 20) + ((size_t)n << 10);
    P[gb + m] = f2bf(a);
  }
}

// ---------------------------------------------------------------------------
// Small helpers: f32 -> bf16 weight conversion, V transpose for attn@v B-side
// ---------------------------------------------------------------------------
__global__ void k_cvt(const float* __restrict__ x,
                      unsigned short* __restrict__ y, int n)
{
  const int i = blockIdx.x * 256 + threadIdx.x;
  if (i < n) y[i] = f2bf(x[i]);
}

__global__ void k_vT(const unsigned short* __restrict__ qkv,
                     unsigned short* __restrict__ vT)
{
  const int idx = blockIdx.x * 256 + threadIdx.x;   // [z][d][m]
  if (idx >= Bn * Hh * HD * Nn) return;
  const int m = idx & 1023;
  const int d = (idx >> 10) & 63;
  const int z = idx >> 16;
  const int b = z / Hh, g = z % Hh;
  vT[idx] = qkv[((size_t)(b * Nn + m)) * (3 * Cc) + 2 * Cc + g * HD + d];
}

// ---------------------------------------------------------------------------
extern "C" void kernel_launch(void* const* d_in, const int* in_sizes, int n_in,
                              void* d_out, int out_size, void* d_ws,
                              size_t ws_size, hipStream_t stream)
{
  (void)in_sizes; (void)n_in; (void)out_size; (void)ws_size;
  const float* x      = (const float*)d_in[0];
  const float* w_qkv  = (const float*)d_in[1];
  const float* b_qkv  = (const float*)d_in[2];
  const float* w_proj = (const float*)d_in[3];
  const float* b_proj = (const float*)d_in[4];
  const float* w_fc1  = (const float*)d_in[5];
  const float* b_fc1  = (const float*)d_in[6];
  const float* w_fc2  = (const float*)d_in[7];
  const float* b_fc2  = (const float*)d_in[8];
  const float* t_bef  = (const float*)d_in[9];
  const float* t_aft  = (const float*)d_in[10];
  const float* g1     = (const float*)d_in[11];
  const float* be1    = (const float*)d_in[12];
  const float* g2     = (const float*)d_in[13];
  const float* be2    = (const float*)d_in[14];
  const float* dw_w   = (const float*)d_in[15];
  const float* dw_b   = (const float*)d_in[16];
  const float* gm     = (const float*)d_in[17];
  const float* bm     = (const float*)d_in[18];

  char* wp_ = (char*)d_ws;
  auto alloc = [&](size_t bytes) -> char* {
    char* p = wp_;
    wp_ += (bytes + 255) & ~(size_t)255;
    return p;
  };
  unsigned short* h1   = (unsigned short*)alloc((size_t)TOK * Cc * 2);
  unsigned short* wqB  = (unsigned short*)alloc((size_t)3 * Cc * Cc * 2);
  unsigned short* wpB  = (unsigned short*)alloc((size_t)Cc * Cc * 2);
  unsigned short* wf1B = (unsigned short*)alloc((size_t)DFF * Cc * 2);
  unsigned short* wf2B = (unsigned short*)alloc((size_t)Cc * DFF * 2);
  unsigned short* qkvB = (unsigned short*)alloc((size_t)TOK * 3 * Cc * 2);
  unsigned short* sc   = (unsigned short*)alloc((size_t)Bn * Hh * Nn * Nn * 2);
  unsigned short* pr   = (unsigned short*)alloc((size_t)Bn * Hh * Nn * Nn * 2);
  unsigned short* vT   = (unsigned short*)alloc((size_t)Bn * Hh * HD * Nn * 2);
  unsigned short* ao   = (unsigned short*)alloc((size_t)TOK * Cc * 2);
  float*          x1   = (float*)alloc((size_t)TOK * Cc * 4);
  float*          h2   = (float*)alloc((size_t)TOK * Cc * 4);
  unsigned short* hm   = (unsigned short*)alloc((size_t)TOK * Cc * 2);
  unsigned short* g1o  = (unsigned short*)alloc((size_t)TOK * DFF * 2);

  // weight conversion fp32 -> bf16
  k_cvt<<<(3 * Cc * Cc + 255) / 256, 256, 0, stream>>>(w_qkv, wqB, 3 * Cc * Cc);
  k_cvt<<<(Cc * Cc + 255) / 256, 256, 0, stream>>>(w_proj, wpB, Cc * Cc);
  k_cvt<<<(DFF * Cc + 255) / 256, 256, 0, stream>>>(w_fc1, wf1B, DFF * Cc);
  k_cvt<<<(Cc * DFF + 255) / 256, 256, 0, stream>>>(w_fc2, wf2B, Cc * DFF);

  // LN1 -> bf16
  k_ln<unsigned short><<<TOK / 8, 256, 0, stream>>>(x, g1, be1, h1);

  // QKV: [4096,2304] = h1 @ w_qkv^T + b_qkv  -> bf16  (block tile 64x256)
  k_gemm<2, 4, 2, 4, 1><<<dim3(3 * Cc / 256, TOK / 64), 256, 0, stream>>>(
      h1, Cc, wqB, Cc, (void*)qkvB, 3 * Cc, Cc, b_qkv, nullptr, 0, 0.f);

  // v transpose for attn@v B operand
  k_vT<<<(Bn * Hh * HD * Nn + 255) / 256, 256, 0, stream>>>(qkvB, vT);

  // scores = q @ k^T * 1/8 per (b,h)
  k_gemm_scores<<<dim3(Nn / 256, Nn / 64, Bn * Hh), 256, 0, stream>>>(qkvB,
                                                                      sc);

  // head-mix -> softmax -> head-mix
  k_mixsoftmax<<<TOK, 256, 0, stream>>>(sc, t_bef, t_aft, pr);

  // attn @ v, scattered into token-major layout (block tile 128x64)
  k_gemm_av<<<dim3(1, Nn / 128, Bn * Hh), 256, 0, stream>>>(pr, vT, ao);

  // proj + bias + residual(x) -> x1 (f32)
  k_gemm<2, 4, 2, 4, 3><<<dim3(Cc / 256, TOK / 64), 256, 0, stream>>>(
      ao, Cc, wpB, Cc, (void*)x1, Cc, Cc, b_proj, x, Cc, 0.f);

  // LN2 -> f32 (feeds depthwise conv)
  k_ln<float><<<TOK / 8, 256, 0, stream>>>(x1, g2, be2, h2);

  // depthwise conv + LN(gm,bm) -> bf16
  k_convln<<<TOK / 8, 256, 0, stream>>>(h2, dw_w, dw_b, gm, bm, hm);

  // fc1 + bias + exact GELU -> bf16
  k_gemm<2, 4, 2, 4, 4><<<dim3(DFF / 256, TOK / 64), 256, 0, stream>>>(
      hm, Cc, wf1B, Cc, (void*)g1o, DFF, Cc, b_fc1, nullptr, 0, 0.f);

  // fc2 + bias + residual(x1) -> d_out (f32)
  k_gemm<2, 4, 2, 4, 3><<<dim3(Cc / 256, TOK / 64), 256, 0, stream>>>(
      g1o, DFF, wf2B, DFF, d_out, Cc, DFF, b_fc2, x1, Cc, 0.f);
}
